// MoEBase_22909355557543
// MI455X (gfx1250) — compile-verified, matrix operand order checked by reading, not run on hardware
//
#include <hip/hip_runtime.h>
#include <hip/hip_bf16.h>

// ---------------- problem constants ----------------
#define D_MODEL 1024
#define HIDDEN  512
#define NE      32          // experts per group
#define T_TOK   4096        // tokens
#define R_ROWS  (T_TOK * 2) // routed rows per group (top-2)
#define CAP     4096        // max tokens per expert (top-2 -> distinct experts)
#define MAX_TILES 544       // >= sum ceil(cnt_e/16) worst case (512 + 31)

typedef __bf16 bf16_t;
typedef unsigned int u32;
typedef __attribute__((ext_vector_type(16))) __bf16 v16bf;
typedef __attribute__((ext_vector_type(8)))  float  v8f;
typedef __attribute__((ext_vector_type(4)))  u32    u32x4;
typedef __attribute__((ext_vector_type(8)))  u32    u32x8;

// Load the 16-element bf16 A-operand for one lane: a[0..7] = k+0..7 (16B),
// a[8..15] = k+16..23 (16B). p must be 16B aligned (kk%32==0, khi in {0,8}).
__device__ __forceinline__ v16bf load_a16(const bf16_t* __restrict__ p) {
    const u32x4 lo = *reinterpret_cast<const u32x4*>(p);
    const u32x4 hi = *reinterpret_cast<const u32x4*>(p + 16);
    u32x8 w;
    w[0] = lo[0]; w[1] = lo[1]; w[2] = lo[2]; w[3] = lo[3];
    w[4] = hi[0]; w[5] = hi[1]; w[6] = hi[2]; w[7] = hi[3];
    return __builtin_bit_cast(v16bf, w);
}

// ---------------- 1. convert x to bf16 ----------------
__global__ void cvt_x_kernel(const float* __restrict__ x, bf16_t* __restrict__ xb, int n) {
    int i = blockIdx.x * blockDim.x + threadIdx.x;
    if (i < n) xb[i] = (bf16_t)x[i];
}

// ---------------- 2. zero per-call metadata + the shared zero row ----------------
// block = 512 threads
__global__ void zero_meta_kernel(int* __restrict__ counts, int* __restrict__ ntiles,
                                 u32* __restrict__ zrow /* D_MODEL bf16 = 512 u32 */) {
    int i = threadIdx.x;
    if (i < 2 * NE) counts[i] = 0;
    if (i < 2)      ntiles[i] = 0;
    zrow[i] = 0u;
}

// ---------------- 3. router: logits -> softmax -> top2 -> expert lists ----------------
// grid (T_TOK, 2 groups), block 32 (one wave; lane == expert)
__global__ void router_topk_kernel(const float* __restrict__ x,
                                   const float* __restrict__ rw0,
                                   const float* __restrict__ rw1,
                                   float* __restrict__ wflat,   // [2][R_ROWS]
                                   int*   __restrict__ counts,  // [2][NE]
                                   int*   __restrict__ lists) { // [2][NE][CAP]
    const int t    = blockIdx.x;
    const int g    = blockIdx.y;
    const int lane = threadIdx.x;               // 0..31 == expert id
    const float* rw = g ? rw1 : rw0;
    const float* xr = x + (size_t)t * D_MODEL;

    float acc = 0.f;
    for (int d = 0; d < D_MODEL; ++d)
        acc += xr[d] * rw[d * NE + lane];

    // softmax across the wave
    float mx = acc;
    for (int o = 16; o > 0; o >>= 1) mx = fmaxf(mx, __shfl_xor(mx, o, 32));
    float p = __expf(acc - mx);
    float sm = p;
    for (int o = 16; o > 0; o >>= 1) sm += __shfl_xor(sm, o, 32);
    float score = p / sm;

    // top-1 (ties -> smaller index, matching argsort-stable top_k)
    float s1 = score; int i1 = lane;
    for (int o = 16; o > 0; o >>= 1) {
        float os = __shfl_xor(s1, o, 32);
        int   oi = __shfl_xor(i1, o, 32);
        if (os > s1 || (os == s1 && oi < i1)) { s1 = os; i1 = oi; }
    }
    // top-2
    float sc2 = (lane == i1) ? -3.0e38f : score;
    float s2 = sc2; int i2 = lane;
    for (int o = 16; o > 0; o >>= 1) {
        float os = __shfl_xor(s2, o, 32);
        int   oi = __shfl_xor(i2, o, 32);
        if (os > s2 || (os == s2 && oi < i2)) { s2 = os; i2 = oi; }
    }

    if (lane == 0) {
        wflat[(size_t)g * R_ROWS + 2 * t]     = s1;
        wflat[(size_t)g * R_ROWS + 2 * t + 1] = s2;
        int p0 = atomicAdd(&counts[g * NE + i1], 1);
        lists[((size_t)g * NE + i1) * CAP + p0] = 2 * t;
        int p1 = atomicAdd(&counts[g * NE + i2], 1);
        lists[((size_t)g * NE + i2) * CAP + p1] = 2 * t + 1;
    }
}

// ---------------- 4. tile worklist from per-expert counts ----------------
// grid (2 groups), block 32 (lane == expert)
__global__ void build_tiles_kernel(const int* __restrict__ counts,
                                   int* __restrict__ offs,
                                   int* __restrict__ ntiles,
                                   int2* __restrict__ tdesc) {
    const int g = blockIdx.x;
    const int lane = threadIdx.x;
    int cnt = counts[g * NE + lane];

    int v = cnt;                               // inclusive scan of counts
    for (int o = 1; o < 32; o <<= 1) { int u = __shfl_up(v, o, 32); if (lane >= o) v += u; }
    offs[g * NE + lane] = v - cnt;             // exclusive -> grouped row offset

    int tcnt = (cnt + 15) >> 4;
    int tv = tcnt;                             // inclusive scan of tile counts
    for (int o = 1; o < 32; o <<= 1) { int u = __shfl_up(tv, o, 32); if (lane >= o) tv += u; }
    int tbase = tv - tcnt;
    if (lane == 31) ntiles[g] = tv;

    int2* td = tdesc + (size_t)g * MAX_TILES;
    for (int i = 0; i < tcnt; ++i) td[tbase + i] = make_int2(lane, i * 16);
}

// ---------------- 5. stage A: h = silu(X@wg) * (X@wu), bf16 WMMA ----------------
// grid (MAX_TILES, HIDDEN/64), block 128 (4 waves; wave -> 16-col subtile)
__global__ void stage_a_kernel(const bf16_t* __restrict__ xb,      // [T][D] bf16
                               const bf16_t* __restrict__ zrow,    // [D] zeros
                               const int*    __restrict__ list,    // [NE][CAP] (group-sel)
                               const int*    __restrict__ counts,  // [NE]
                               const int*    __restrict__ offs,    // [NE]
                               const int*    __restrict__ ntiles,
                               const int2*   __restrict__ tdesc,
                               const float*  __restrict__ wg,      // [NE][D][H]
                               const float*  __restrict__ wu,      // [NE][D][H]
                               bf16_t*       __restrict__ hbuf) {  // [(R+16)][H] bf16
    const int tile = blockIdx.x;
    if (tile >= *ntiles) return;
    const int2 td    = tdesc[tile];
    const int  e     = td.x;
    const int  start = td.y;
    const int  lane  = threadIdx.x & 31;
    const int  wav   = threadIdx.x >> 5;
    const int  nbase = blockIdx.y * 64 + wav * 16;
    const int  cnt   = counts[e];
    const int  goff  = offs[e];

    // A layout: this lane supplies row m = lane&15, K half per lane>>4
    const int mA   = lane & 15;
    const int khi  = (lane >> 4) << 3;          // 0 or 8
    const int rowA = start + mA;
    const bool validA = rowA < cnt;
    const int entry = validA ? list[(size_t)e * CAP + rowA] : 0;
    const bf16_t* xrow = validA ? (xb + (size_t)(entry >> 1) * D_MODEL) : zrow;

    const float* wgp = wg + (size_t)e * D_MODEL * HIDDEN;
    const float* wup = wu + (size_t)e * D_MODEL * HIDDEN;
    const int ncol = nbase + (lane & 15);       // B/C/D col for this lane

    v8f cg = {}; v8f cu = {};
    for (int kk = 0; kk < D_MODEL; kk += 32) {
        const v16bf a = load_a16(xrow + kk + khi);
        v16bf bg, bu;
#pragma unroll
        for (int v = 0; v < 8; ++v) {
            const int k = kk + ((v & 3) << 1) + khi + ((v >> 2) << 4);
            bg[2 * v]     = (bf16_t)wgp[(size_t)k * HIDDEN + ncol];
            bg[2 * v + 1] = (bf16_t)wgp[(size_t)(k + 1) * HIDDEN + ncol];
            bu[2 * v]     = (bf16_t)wup[(size_t)k * HIDDEN + ncol];
            bu[2 * v + 1] = (bf16_t)wup[(size_t)(k + 1) * HIDDEN + ncol];
        }
        cg = __builtin_amdgcn_wmma_f32_16x16x32_bf16(false, a, false, bg, (short)0, cg, false, false);
        cu = __builtin_amdgcn_wmma_f32_16x16x32_bf16(false, a, false, bu, (short)0, cu, false, false);
    }

#pragma unroll
    for (int v = 0; v < 8; ++v) {
        const int m   = v + ((lane >> 4) << 3);
        const int row = start + m;
        if (row < cnt) {
            const float gv = cg[v], uv = cu[v];
            // silu(g)*u with single v_rcp_f32 (bf16 output tolerates rcp rounding)
            const float sig = __builtin_amdgcn_rcpf(1.f + __expf(-gv));
            const float hv  = gv * sig * uv;
            hbuf[(size_t)(goff + row) * HIDDEN + ncol] = (bf16_t)hv;
        }
    }
}

// ---------------- 6. stage B: ys = (h @ wd) * route_w, scatter to slots ----------------
// grid (MAX_TILES, D_MODEL/64), block 128
__global__ void stage_b_kernel(const bf16_t* __restrict__ hbuf,    // [(R+16)][H] bf16
                               const int*    __restrict__ list,
                               const int*    __restrict__ counts,
                               const int*    __restrict__ offs,
                               const int*    __restrict__ ntiles,
                               const int2*   __restrict__ tdesc,
                               const float*  __restrict__ wd,      // [NE][H][D]
                               const float*  __restrict__ wflat,   // [R] (group-sel)
                               float*        __restrict__ ys) {    // [R][D]
    const int tile = blockIdx.x;
    if (tile >= *ntiles) return;
    const int2 td    = tdesc[tile];
    const int  e     = td.x;
    const int  start = td.y;
    const int  lane  = threadIdx.x & 31;
    const int  wav   = threadIdx.x >> 5;
    const int  nbase = blockIdx.y * 64 + wav * 16;
    const int  cnt   = counts[e];
    const int  goff  = offs[e];

    const int mA  = lane & 15;
    const int khi = (lane >> 4) << 3;
    const bf16_t* hrow = hbuf + (size_t)(goff + start + mA) * HIDDEN;  // slack rows make OOB safe
    const float*  wdp  = wd + (size_t)e * HIDDEN * D_MODEL;
    const int ncol = nbase + (lane & 15);

    v8f c = {};
    for (int kk = 0; kk < HIDDEN; kk += 32) {
        const v16bf a = load_a16(hrow + kk + khi);
        v16bf b;
#pragma unroll
        for (int v = 0; v < 8; ++v) {
            const int k = kk + ((v & 3) << 1) + khi + ((v >> 2) << 4);
            b[2 * v]     = (bf16_t)wdp[(size_t)k * D_MODEL + ncol];
            b[2 * v + 1] = (bf16_t)wdp[(size_t)(k + 1) * D_MODEL + ncol];
        }
        c = __builtin_amdgcn_wmma_f32_16x16x32_bf16(false, a, false, b, (short)0, c, false, false);
    }

#pragma unroll
    for (int v = 0; v < 8; ++v) {
        const int m   = v + ((lane >> 4) << 3);
        const int row = start + m;
        if (row < cnt) {
            const int ent = list[(size_t)e * CAP + row];       // slot = t*2+k
            ys[(size_t)ent * D_MODEL + ncol] = c[v] * wflat[ent];
        }
    }
}

// ---------------- 7. combine: out (=|+=) 0.5*(ys[2t] + ys[2t+1]) ----------------
__global__ void combine_kernel(const float* __restrict__ ys, float* __restrict__ out, int add) {
    const long i = (long)blockIdx.x * blockDim.x + threadIdx.x;  // i = t*D + d
    const long t = i / D_MODEL;
    const long d = i - t * D_MODEL;
    const float v = 0.5f * (ys[(2 * t) * (long)D_MODEL + d] +
                            ys[(2 * t + 1) * (long)D_MODEL + d]);
    if (add) out[i] += v; else out[i] = v;
}

// ---------------- host launch ----------------
extern "C" void kernel_launch(void* const* d_in, const int* in_sizes, int n_in,
                              void* d_out, int out_size, void* d_ws, size_t ws_size,
                              hipStream_t stream) {
    const float* x   = (const float*)d_in[0];
    const float* rw0 = (const float*)d_in[1];
    const float* wg0 = (const float*)d_in[2];
    const float* wu0 = (const float*)d_in[3];
    const float* wd0 = (const float*)d_in[4];
    const float* rw1 = (const float*)d_in[5];
    const float* wg1 = (const float*)d_in[6];
    const float* wu1 = (const float*)d_in[7];
    const float* wd1 = (const float*)d_in[8];
    float* out = (float*)d_out;

    // workspace carve-up (256B aligned)
    char* base = (char*)d_ws;
    size_t off = 0;
    auto carve = [&](size_t bytes) { size_t o = off; off = (off + bytes + 255) & ~(size_t)255; return o; };
    bf16_t* xb     = (bf16_t*)(base + carve((size_t)T_TOK * D_MODEL * sizeof(bf16_t)));
    bf16_t* zrow   = (bf16_t*)(base + carve((size_t)D_MODEL * sizeof(bf16_t)));
    float*  wflat  = (float*) (base + carve((size_t)2 * R_ROWS * sizeof(float)));
    int*    counts = (int*)   (base + carve((size_t)2 * NE * sizeof(int)));
    int*    offs   = (int*)   (base + carve((size_t)2 * NE * sizeof(int)));
    int*    ntiles = (int*)   (base + carve((size_t)2 * sizeof(int)));
    int*    lists  = (int*)   (base + carve((size_t)2 * NE * CAP * sizeof(int)));
    int2*   tdesc  = (int2*)  (base + carve((size_t)2 * MAX_TILES * sizeof(int2)));
    bf16_t* hbuf   = (bf16_t*)(base + carve((size_t)(R_ROWS + 16) * HIDDEN * sizeof(bf16_t)));
    float*  ys     = (float*) (base + carve((size_t)R_ROWS * D_MODEL * sizeof(float)));
    (void)ws_size; (void)n_in; (void)in_sizes; (void)out_size;

    // 1. x -> bf16
    cvt_x_kernel<<<(T_TOK * D_MODEL) / 256, 256, 0, stream>>>(x, xb, T_TOK * D_MODEL);
    // 2. zero per-call metadata + zero row
    zero_meta_kernel<<<1, 512, 0, stream>>>(counts, ntiles, (u32*)zrow);
    // 3. routing + expert lists (both groups)
    router_topk_kernel<<<dim3(T_TOK, 2), 32, 0, stream>>>(x, rw0, rw1, wflat, counts, lists);
    // 4. tile worklists
    build_tiles_kernel<<<2, 32, 0, stream>>>(counts, offs, ntiles, tdesc);

    // 5/6/7: per group, reuse hbuf/ys sequentially
    const float* wgs[2] = {wg0, wg1};
    const float* wus[2] = {wu0, wu1};
    const float* wds[2] = {wd0, wd1};
    for (int g = 0; g < 2; ++g) {
        const int*  lst = lists + (size_t)g * NE * CAP;
        const int*  cnt = counts + g * NE;
        const int*  ofg = offs + g * NE;
        const int*  ntg = ntiles + g;
        const int2* tdg = tdesc + (size_t)g * MAX_TILES;

        stage_a_kernel<<<dim3(MAX_TILES, HIDDEN / 64), 128, 0, stream>>>(
            xb, zrow, lst, cnt, ofg, ntg, tdg, wgs[g], wus[g], hbuf);
        stage_b_kernel<<<dim3(MAX_TILES, D_MODEL / 64), 128, 0, stream>>>(
            hbuf, lst, cnt, ofg, ntg, tdg, wds[g], wflat + (size_t)g * R_ROWS, ys);
        combine_kernel<<<(T_TOK * D_MODEL) / 256, 256, 0, stream>>>(ys, out, g);
    }
}
